// MultiHead_21406117003599
// MI455X (gfx1250) — matrix operand order repaired and checked by reading, not compiled, not run on hardware
//
#include <hip/hip_runtime.h>
#include <hip/hip_bf16.h>
#include <stdint.h>

#define BATCH 4
#define SEQ   2048
#define NEMBD 1024
#define NHEAD 16
#define HSIZE 64
#define ROWS  (BATCH * SEQ)   // 8192 flat rows of x / att / out

typedef unsigned short u16;
typedef __attribute__((ext_vector_type(16))) __bf16 v16bf;
typedef __attribute__((ext_vector_type(8)))  float  v8f;

union FragU { v16bf v; uint4 q[2]; };
union Pack8 { uint4 q; u16 u[8]; };
union Pack4 { uint2 d; u16 u[4]; };

__device__ __forceinline__ u16 f2bf(float f) {
  unsigned int u = __float_as_uint(f);
  u += 0x7FFFu + ((u >> 16) & 1u);   // round-to-nearest-even truncation
  return (u16)(u >> 16);
}

__device__ __forceinline__ v8f zero8() {
  v8f z = {0.f, 0.f, 0.f, 0.f, 0.f, 0.f, 0.f, 0.f};
  return z;
}

// --- CDNA5 async global->LDS copy (ASYNCcnt path), 16B per lane ------------
__device__ __forceinline__ void async_b128(void* lds, const void* gaddr) {
  asm volatile("global_load_async_to_lds_b128 %0, %1, off"
               :: "v"((unsigned)(uintptr_t)lds), "v"(gaddr)
               : "memory");
}
__device__ __forceinline__ void wait_async0() {
  asm volatile("s_wait_asynccnt 0x0" ::: "memory");
}
__device__ __forceinline__ void wait_ds0() {
  asm volatile("s_wait_dscnt 0x0" ::: "memory");
}

// A fragment 16x32 bf16 from a row-major [m][k] source; `row` points at this
// lane's M row (m = lane&15). ISA layout: lanes 0-15 hold K {0..7,16..23},
// lanes 16-31 hold K {8..15,24..31}.
__device__ __forceinline__ v16bf load_a(const u16* row, int kbase, int lane) {
  const int b = (lane & 16) ? 8 : 0;
  FragU f;
  f.q[0] = *(const uint4*)(row + kbase + b);
  f.q[1] = *(const uint4*)(row + kbase + b + 16);
  return f.v;
}

// B fragment 32x16 bf16 from an [n][k]-major source; `row` points at this
// lane's N row (n = lane&15). Lanes 0-15 hold K {0..7,16..23}, lanes 16-31 K {8..15,24..31}.
__device__ __forceinline__ v16bf load_b(const u16* row, int kbase, int lane) {
  const int koff = (lane & 16) ? 8 : 0;
  FragU f;
  f.q[0] = *(const uint4*)(row + kbase + koff);
  f.q[1] = *(const uint4*)(row + kbase + koff + 16);
  return f.v;
}

__device__ __forceinline__ v8f wmma_bf16(v16bf a, v16bf b, v8f c) {
  return __builtin_amdgcn_wmma_f32_16x16x32_bf16(false, a, false, b,
                                                 (short)0, c, false, false);
}

// ---------------------------------------------------------------- converts --
__global__ void k_cast_bf16(const float* __restrict__ src,
                            u16* __restrict__ dst, int n) {
  int i = (blockIdx.x * blockDim.x + threadIdx.x) * 4;
  if (i + 3 < n) {
    float4 v = *(const float4*)(src + i);
    Pack4 p;
    p.u[0] = f2bf(v.x); p.u[1] = f2bf(v.y);
    p.u[2] = f2bf(v.z); p.u[3] = f2bf(v.w);
    *(uint2*)(dst + i) = p.d;
  }
}

// W[h][e][d] (fp32) -> WT[h][d][e] (bf16)
__global__ void k_transpose_hed(const float* __restrict__ W,
                                u16* __restrict__ WT) {
  int idx = blockIdx.x * blockDim.x + threadIdx.x;  // over H*D*E = 2^20
  int e = idx & (NEMBD - 1);
  int d = (idx >> 10) & (HSIZE - 1);
  int h = idx >> 16;
  WT[idx] = f2bf(W[((size_t)h * NEMBD + e) * HSIZE + d]);
}

// Wproj[k][n] (fp32) -> WT[n][k] (bf16)
__global__ void k_transpose_proj(const float* __restrict__ W,
                                 u16* __restrict__ WT) {
  int idx = blockIdx.x * blockDim.x + threadIdx.x;  // over 2^20
  int n = idx >> 10, k = idx & (NEMBD - 1);
  WT[idx] = f2bf(W[(size_t)k * NEMBD + n]);
}

// -------------------------------------------------------------------- QKV ---
// grid (ROWS/128, NHEAD), block 256 (8 waves). Each wave: 16 rows x 64 cols
// for Q, K and V of one head; K-loop over E in steps of 32, double-buffered
// async staging. V is written transposed (VT[b][h][d][t]) for the PV stage.
__global__ __launch_bounds__(256) void k_qkv(
    const u16* __restrict__ xb, const u16* __restrict__ WqT,
    const u16* __restrict__ WkT, const u16* __restrict__ WvT,
    u16* __restrict__ Q, u16* __restrict__ K, u16* __restrict__ VT) {
  __shared__ __align__(16) u16 xs[2][128][40];     // 32 k + 8 pad
  __shared__ __align__(16) u16 wsm[2][3][64][40];  // [buf][qkv][d][k]

  const int tid = threadIdx.x;
  const int lane = tid & 31;
  const int w = tid >> 5;
  const int lrow = lane & 15;
  const int hi8 = (lane & 16) ? 8 : 0;
  const int h = blockIdx.y;
  const int m0 = blockIdx.x * 128;

  const u16* Wt0 = WqT + (size_t)h * HSIZE * NEMBD;
  const u16* Wt1 = WkT + (size_t)h * HSIZE * NEMBD;
  const u16* Wt2 = WvT + (size_t)h * HSIZE * NEMBD;

  const int sr = tid >> 2, sseg = (tid & 3) * 8;   // staging coords

  auto stage = [&](int buf, int kk) {
    const int k0 = kk * 32;
    async_b128(&xs[buf][sr][sseg],      &xb[(size_t)(m0 + sr)      * NEMBD + k0 + sseg]);
    async_b128(&xs[buf][sr + 64][sseg], &xb[(size_t)(m0 + sr + 64) * NEMBD + k0 + sseg]);
    async_b128(&wsm[buf][0][sr][sseg],  &Wt0[(size_t)sr * NEMBD + k0 + sseg]);
    async_b128(&wsm[buf][1][sr][sseg],  &Wt1[(size_t)sr * NEMBD + k0 + sseg]);
    async_b128(&wsm[buf][2][sr][sseg],  &Wt2[(size_t)sr * NEMBD + k0 + sseg]);
  };

  v8f accQ[4], accK[4], accV[4];
  for (int nt = 0; nt < 4; ++nt) { accQ[nt] = zero8(); accK[nt] = zero8(); accV[nt] = zero8(); }

  stage(0, 0);
  wait_async0();
  __syncthreads();

  for (int kk = 0; kk < NEMBD / 32; ++kk) {
    const int cur = kk & 1;
    if (kk + 1 < NEMBD / 32) stage(cur ^ 1, kk + 1);

    // Software-pipelined fragment consumption: each WMMA eats a fragment
    // loaded one group earlier, so partial s_wait_dscnt can cover it.
    const v16bf a = load_a(&xs[cur][w * 16 + lrow][0], 0, lane);
    v16bf bq[4], bk[4], bv[4];
    for (int nt = 0; nt < 4; ++nt)
      bq[nt] = load_b(&wsm[cur][0][nt * 16 + lrow][0], 0, lane);
    for (int nt = 0; nt < 4; ++nt) {
      bk[nt] = load_b(&wsm[cur][1][nt * 16 + lrow][0], 0, lane);
      accQ[nt] = wmma_bf16(a, bq[nt], accQ[nt]);
    }
    for (int nt = 0; nt < 4; ++nt) {
      bv[nt] = load_b(&wsm[cur][2][nt * 16 + lrow][0], 0, lane);
      accK[nt] = wmma_bf16(a, bk[nt], accK[nt]);
    }
    for (int nt = 0; nt < 4; ++nt)
      accV[nt] = wmma_bf16(a, bv[nt], accV[nt]);

    wait_async0();
    __syncthreads();
  }

  const int m_base = m0 + w * 16 + hi8;           // 8 consecutive flat rows
  const int bb = m_base >> 11, t_base = m_base & (SEQ - 1);
  const size_t bh = (size_t)(bb * NHEAD + h);
  for (int nt = 0; nt < 4; ++nt) {
    const int d = nt * 16 + lrow;
    for (int i = 0; i < 8; ++i) {
      Q[(bh * SEQ + t_base + i) * HSIZE + d] = f2bf(accQ[nt][i]);
      K[(bh * SEQ + t_base + i) * HSIZE + d] = f2bf(accK[nt][i]);
    }
    // V transposed: 8 consecutive t per lane -> one 16B store
    Pack8 pv;
    for (int i = 0; i < 8; ++i) pv.u[i] = f2bf(accV[nt][i]);
    *(uint4*)&VT[(bh * HSIZE + d) * SEQ + t_base] = pv.q;
  }
}

// ------------------------------------------------------------ flash attn ----
// grid (SEQ/64, NHEAD, BATCH), block 128 (4 waves). 64 query rows per block,
// K/V streamed in causal S-blocks of 64 with online softmax; double-buffered
// async tile staging; one workgroup barrier per S-block.
__global__ __launch_bounds__(128) void k_flash(
    const u16* __restrict__ Q, const u16* __restrict__ K,
    const u16* __restrict__ VT, u16* __restrict__ att) {
  __shared__ __align__(16) u16 Ks[2][64][72];    // [buf][s][d]
  __shared__ __align__(16) u16 Vt[2][64][72];    // [buf][d][s]
  __shared__ __align__(16) u16 Ps[4][16][72];    // per-wave P tile [m][s]

  const int tid = threadIdx.x, lane = tid & 31, w = tid >> 5;
  const int lrow = lane & 15;
  const int hi8 = (lane & 16) ? 8 : 0;
  const int t0 = blockIdx.x * 64;
  const int h = blockIdx.y, b = blockIdx.z;
  const size_t bh = (size_t)(b * NHEAD + h);
  const u16* Qb  = Q  + bh * SEQ * HSIZE;
  const u16* Kb  = K  + bh * SEQ * HSIZE;
  const u16* VTb = VT + bh * HSIZE * SEQ;

  const int sr = tid >> 1, shalf = (tid & 1) * 32;  // staging coords

  auto stage = [&](int buf, int s0) {
    for (int seg = 0; seg < 2; ++seg) {
      const int c = shalf + seg * 8;
      async_b128(&Ks[buf][sr][c], &Kb[(size_t)(s0 + sr) * HSIZE + c]);
      async_b128(&Vt[buf][sr][c], &VTb[(size_t)sr * SEQ + s0 + c]);
    }
  };

  const int trow = t0 + w * 16 + lrow;
  const v16bf qf0 = load_a(&Qb[(size_t)trow * HSIZE], 0, lane);
  const v16bf qf1 = load_a(&Qb[(size_t)trow * HSIZE], 32, lane);

  v8f oacc[4];
  float mrun[8], lrun[8];
  for (int nt = 0; nt < 4; ++nt) oacc[nt] = zero8();
  const float NEGINF = -__builtin_inff();
  for (int i = 0; i < 8; ++i) { mrun[i] = NEGINF; lrun[i] = 0.f; }

  const int nblk = blockIdx.x + 1;  // s0 = 0 .. t0 step 64 (causal)
  stage(0, 0);
  wait_async0();
  __syncthreads();

  for (int blk = 0; blk < nblk; ++blk) {
    const int s0 = blk * 64;
    const int cur = blk & 1;
    if (blk + 1 < nblk) stage(cur ^ 1, (blk + 1) * 64);

    // scores S = Q K^T  (B-frag [n=s][k=d] == Ks rows)
    v8f s[4];
    {
      v16bf b0[4], b1[4];
      for (int nt = 0; nt < 4; ++nt) {
        const u16* krow = &Ks[cur][nt * 16 + lrow][0];
        b0[nt] = load_b(krow, 0, lane);
        b1[nt] = load_b(krow, 32, lane);
      }
      for (int nt = 0; nt < 4; ++nt) {
        v8f c = zero8();
        c = wmma_bf16(qf0, b0[nt], c);
        s[nt] = wmma_bf16(qf1, b1[nt], c);
      }
    }
    // scale + causal mask
    const bool diag = (s0 + 63 > t0 + w * 16);
    for (int nt = 0; nt < 4; ++nt)
      for (int i = 0; i < 8; ++i) {
        float v = s[nt][i] * 0.125f;   // 1/sqrt(64)
        if (diag) {
          const int mg = t0 + w * 16 + hi8 + i;
          const int ng = s0 + nt * 16 + lrow;
          if (ng > mg) v = NEGINF;
        }
        s[nt][i] = v;
      }
    // online softmax; row i lives across 16 lanes of one half-wave
    for (int i = 0; i < 8; ++i) {
      float rm = fmaxf(fmaxf(s[0][i], s[1][i]), fmaxf(s[2][i], s[3][i]));
      for (int off = 1; off < 16; off <<= 1)
        rm = fmaxf(rm, __shfl_xor(rm, off, 32));
      const float nm = fmaxf(mrun[i], rm);
      const float fac = __expf(mrun[i] - nm);
      mrun[i] = nm;
      float rs = 0.f;
      for (int nt = 0; nt < 4; ++nt) {
        const float p = __expf(s[nt][i] - nm);
        s[nt][i] = p;
        rs += p;
      }
      for (int off = 1; off < 16; off <<= 1) rs += __shfl_xor(rs, off, 32);
      lrun[i] = lrun[i] * fac + rs;
      for (int nt = 0; nt < 4; ++nt) oacc[nt][i] = oacc[nt][i] * fac;
    }
    // D-layout -> A-layout relayout of P through per-wave LDS scratch.
    // Same-wave cross-lane exchange: LDS is in-order per wave; wait + fence.
    for (int nt = 0; nt < 4; ++nt)
      for (int i = 0; i < 8; ++i)
        Ps[w][hi8 + i][nt * 16 + lrow] = f2bf(s[nt][i]);
    wait_ds0();
    __builtin_amdgcn_wave_barrier();

    // O += P V  (B-frag [n=d][k=s] == Vt rows)
    {
      const v16bf pf0 = load_a(&Ps[w][lrow][0], 0, lane);
      const v16bf pf1 = load_a(&Ps[w][lrow][0], 32, lane);
      v16bf b0[4], b1[4];
      for (int nt = 0; nt < 4; ++nt) {
        const u16* vrow = &Vt[cur][nt * 16 + lrow][0];
        b0[nt] = load_b(vrow, 0, lane);
        b1[nt] = load_b(vrow, 32, lane);
      }
      for (int nt = 0; nt < 4; ++nt) {
        oacc[nt] = wmma_bf16(pf0, b0[nt], oacc[nt]);
        oacc[nt] = wmma_bf16(pf1, b1[nt], oacc[nt]);
      }
    }

    wait_async0();
    __syncthreads();
  }

  // normalize and write concat-head layout att[b*SEQ+t][h*64+d]
  for (int nt = 0; nt < 4; ++nt) {
    const int d = nt * 16 + lrow;
    for (int i = 0; i < 8; ++i) {
      const int t = t0 + w * 16 + hi8 + i;
      const float o = oacc[nt][i] / lrun[i];
      att[(size_t)(b * SEQ + t) * NEMBD + h * HSIZE + d] = f2bf(o);
    }
  }
}

// ------------------------------------------------------------- out proj -----
// grid (ROWS/128, NEMBD/64), block 256 (8 waves). out = att @ Wproj + b.
__global__ __launch_bounds__(256) void k_proj(
    const u16* __restrict__ att, const u16* __restrict__ WpT,
    const float* __restrict__ bias, float* __restrict__ out) {
  __shared__ __align__(16) u16 As[2][128][40];
  __shared__ __align__(16) u16 Bs[2][64][40];
  const int tid = threadIdx.x, lane = tid & 31, w = tid >> 5;
  const int lrow = lane & 15;
  const int hi8 = (lane & 16) ? 8 : 0;
  const int m0 = blockIdx.x * 128, n0 = blockIdx.y * 64;

  const int sr = tid >> 2, sseg = (tid & 3) * 8;

  auto stage = [&](int buf, int kk) {
    const int k0 = kk * 32;
    async_b128(&As[buf][sr][sseg],      &att[(size_t)(m0 + sr)      * NEMBD + k0 + sseg]);
    async_b128(&As[buf][sr + 64][sseg], &att[(size_t)(m0 + sr + 64) * NEMBD + k0 + sseg]);
    async_b128(&Bs[buf][sr][sseg],      &WpT[(size_t)(n0 + sr)      * NEMBD + k0 + sseg]);
  };

  v8f acc[4];
  for (int nt = 0; nt < 4; ++nt) acc[nt] = zero8();

  stage(0, 0);
  wait_async0();
  __syncthreads();

  for (int kk = 0; kk < NEMBD / 32; ++kk) {
    const int cur = kk & 1;
    if (kk + 1 < NEMBD / 32) stage(cur ^ 1, kk + 1);

    const v16bf a = load_a(&As[cur][w * 16 + lrow][0], 0, lane);
    v16bf bf[4];
    for (int nt = 0; nt < 4; ++nt)
      bf[nt] = load_b(&Bs[cur][nt * 16 + lrow][0], 0, lane);
    for (int nt = 0; nt < 4; ++nt)
      acc[nt] = wmma_bf16(a, bf[nt], acc[nt]);

    wait_async0();
    __syncthreads();
  }

  for (int nt = 0; nt < 4; ++nt) {
    const int n = n0 + nt * 16 + lrow;
    const float bv = bias[n];
    for (int i = 0; i < 8; ++i) {
      const int m = m0 + w * 16 + hi8 + i;
      out[(size_t)m * NEMBD + n] = acc[nt][i] + bv;
    }
  }
}

// ------------------------------------------------------------------ launch --
extern "C" void kernel_launch(void* const* d_in, const int* in_sizes, int n_in,
                              void* d_out, int out_size, void* d_ws,
                              size_t ws_size, hipStream_t stream) {
  (void)in_sizes; (void)n_in; (void)out_size; (void)ws_size;
  const float* x  = (const float*)d_in[0];
  const float* Wq = (const float*)d_in[1];
  const float* Wk = (const float*)d_in[2];
  const float* Wv = (const float*)d_in[3];
  const float* Wp = (const float*)d_in[4];
  const float* bp = (const float*)d_in[5];
  float* out = (float*)d_out;

  char* p = (char*)d_ws;
  u16* xb  = (u16*)p; p += (size_t)ROWS * NEMBD * 2;                 // 16 MB
  u16* WqT = (u16*)p; p += (size_t)NHEAD * HSIZE * NEMBD * 2;        //  2 MB
  u16* WkT = (u16*)p; p += (size_t)NHEAD * HSIZE * NEMBD * 2;
  u16* WvT = (u16*)p; p += (size_t)NHEAD * HSIZE * NEMBD * 2;
  u16* WpT = (u16*)p; p += (size_t)NEMBD * NEMBD * 2;                //  2 MB
  u16* Qd  = (u16*)p; p += (size_t)BATCH * NHEAD * SEQ * HSIZE * 2;  // 16 MB
  u16* Kd  = (u16*)p; p += (size_t)BATCH * NHEAD * SEQ * HSIZE * 2;
  u16* VTd = (u16*)p; p += (size_t)BATCH * NHEAD * SEQ * HSIZE * 2;
  u16* attb = (u16*)p;                                               // 16 MB

  k_cast_bf16<<<(ROWS * NEMBD) / 1024, 256, 0, stream>>>(x, xb, ROWS * NEMBD);
  k_transpose_hed<<<(NHEAD * HSIZE * NEMBD) / 256, 256, 0, stream>>>(Wq, WqT);
  k_transpose_hed<<<(NHEAD * HSIZE * NEMBD) / 256, 256, 0, stream>>>(Wk, WkT);
  k_transpose_hed<<<(NHEAD * HSIZE * NEMBD) / 256, 256, 0, stream>>>(Wv, WvT);
  k_transpose_proj<<<(NEMBD * NEMBD) / 256, 256, 0, stream>>>(Wp, WpT);

  k_qkv<<<dim3(ROWS / 128, NHEAD), 256, 0, stream>>>(xb, WqT, WkT, WvT,
                                                     Qd, Kd, VTd);
  k_flash<<<dim3(SEQ / 64, NHEAD, BATCH), 128, 0, stream>>>(Qd, Kd, VTd, attb);
  k_proj<<<dim3(ROWS / 128, NEMBD / 64), 256, 0, stream>>>(attb, WpT, bp, out);
}
